// MoLoraLayer_19061064860146
// MI455X (gfx1250) — matrix operand order repaired and checked by reading, not compiled, not run on hardware
//
#include <hip/hip_runtime.h>

typedef __bf16 bf16_t;
typedef bf16_t v16bf __attribute__((ext_vector_type(16)));
typedef float  v8f   __attribute__((ext_vector_type(8)));

union FragU { uint4 q[2]; v16bf v; };

constexpr int D_DIM   = 1024;
constexpr int E_NUM   = 8;
constexpr int R_DIM   = 16;
constexpr int TILE_M  = 16;
constexpr int THREADS = 128;            // 4 waves (wave32)
constexpr int KB_X    = D_DIM / 32;     // 32 K-blocks over D

// d_ws layout (dword offsets): bf16 hi/lo fragment planes
constexpr int WG_HI = 0;                 // 32 kb * 256 dw
constexpr int WG_LO = 8192;
constexpr int A_HI  = 16384;             // (8*32) kb * 256 dw
constexpr int A_LO  = 81920;
constexpr int B_HI  = 147456;            // (64*4) blocks * 256 dw
constexpr int B_LO  = 212992;
constexpr int WS_DW = 278528;            // ~1.06 MB

// LDS layout (dword offsets)
constexpr int XF_HI  = 0;      // 32 kb * 32 lanes * 8 dw
constexpr int XF_LO  = 8192;
constexpr int HWF_HI = 16384;  // 4 kb * 32 * 8 dw
constexpr int HWF_LO = 17408;
constexpr int GPART  = 18432;  // 4 waves * 16*16 f32
constexpr int CW     = 19456;  // 16*8 f32
constexpr int LDS_DW = 19584;  // 78336 B

// ---- Pack a row-major f32 weight matrix into bf16 hi/lo B-fragment planes ----
// src[k][c] with row stride ld; frag block = nt*kblocks + kb; column c = nt*16 + n.
// Per-lane 16 bf16: n = lane&15, K = kb*32 + (lane>>4)*8 + (d>>2)*16 + (d&3)*2 + [0,1] for dword d.
__global__ void pack_weights_kernel(const float* __restrict__ src,
                                    unsigned int* __restrict__ dhi,
                                    unsigned int* __restrict__ dlo,
                                    int ld, int kblocks, int ntiles, int nv) {
    int idx = blockIdx.x * 256 + threadIdx.x;
    int total = ntiles * kblocks * 256;
    if (idx >= total) return;
    int d    = idx & 7;
    int lane = (idx >> 3) & 31;
    int blk  = idx >> 8;
    int kb   = blk % kblocks;
    int nt   = blk / kblocks;
    int n    = lane & 15;
    int k0   = kb * 32 + ((lane >> 4) << 3) + ((d >> 2) << 4) + ((d & 3) << 1);
    float f0 = 0.0f, f1 = 0.0f;
    if (n < nv) {
        long c = (long)nt * 16 + n;
        f0 = src[(long)k0 * ld + c];
        f1 = src[(long)(k0 + 1) * ld + c];
    }
    unsigned int u0 = __float_as_uint(f0), u1 = __float_as_uint(f1);
    dhi[idx] = (u0 >> 16) | (u1 & 0xffff0000u);
    float r0 = f0 - __uint_as_float(u0 & 0xffff0000u);
    float r1 = f1 - __uint_as_float(u1 & 0xffff0000u);
    dlo[idx] = (__float_as_uint(r0) >> 16) | (__float_as_uint(r1) & 0xffff0000u);
}

__device__ inline v8f wmma3(const FragU& ahi, const FragU& alo,
                            const FragU& bhi, const FragU& blo, v8f acc) {
    // small terms first, then hi*hi (all f32 accumulate)
    acc = __builtin_amdgcn_wmma_f32_16x16x32_bf16(false, alo.v, false, bhi.v,
                                                  (short)0, acc, false, false);
    acc = __builtin_amdgcn_wmma_f32_16x16x32_bf16(false, ahi.v, false, blo.v,
                                                  (short)0, acc, false, false);
    acc = __builtin_amdgcn_wmma_f32_16x16x32_bf16(false, ahi.v, false, bhi.v,
                                                  (short)0, acc, false, false);
    return acc;
}

__global__ __launch_bounds__(THREADS) void molora_bf16x3_kernel(
    const float* __restrict__ x,            // [T, 1024] f32
    const unsigned int* __restrict__ ws,    // packed fragment planes
    float* __restrict__ out)                // [T, 1024] f32
{
    extern __shared__ unsigned int smem_u[];
    const int  tid   = threadIdx.x;
    const int  wave  = tid >> 5;
    const int  lane  = tid & 31;
    const int  mn    = lane & 15;          // M (A rows) or N (B/C cols)
    const int  mhigh = (lane >> 4) << 3;   // +8 row offset for C/D high half
    const long tok0  = (long)blockIdx.x * TILE_M;

    // ---------- Phase A: stream x tile, split hi/lo bf16, store A-frag layout ----------
    {
        const uint4* src = (const uint4*)(x + tok0 * D_DIM);
        #pragma unroll
        for (int i = 0; i < (TILE_M * D_DIM / 4) / THREADS; ++i) {
            int f4 = i * THREADS + tid;          // float4 index, coalesced
            int m  = f4 >> 8;                    // token row
            int k  = (f4 & 255) << 2;            // k base (mult of 4)
            uint4 w = src[f4];
            unsigned int hiA = (w.x >> 16) | (w.y & 0xffff0000u);
            unsigned int hiB = (w.z >> 16) | (w.w & 0xffff0000u);
            float r0 = __uint_as_float(w.x) - __uint_as_float(w.x & 0xffff0000u);
            float r1 = __uint_as_float(w.y) - __uint_as_float(w.y & 0xffff0000u);
            float r2 = __uint_as_float(w.z) - __uint_as_float(w.z & 0xffff0000u);
            float r3 = __uint_as_float(w.w) - __uint_as_float(w.w & 0xffff0000u);
            unsigned int loA = (__float_as_uint(r0) >> 16) | (__float_as_uint(r1) & 0xffff0000u);
            unsigned int loB = (__float_as_uint(r2) >> 16) | (__float_as_uint(r3) & 0xffff0000u);
            // A-fragment position: lane = m + 16*h, h = bit3 of k%32; p = (k&7) + 8*bit4
            int kl = k & 31;
            int h  = (kl >> 3) & 1;
            int p  = (kl & 7) + ((kl >> 4) & 1) * 8;
            int dw = (((k >> 5) * 32) + m + (h << 4)) * 8 + (p >> 1);  // even
            uint2 hv; hv.x = hiA; hv.y = hiB;
            uint2 lv; lv.x = loA; lv.y = loB;
            *(uint2*)(smem_u + XF_HI + dw) = hv;
            *(uint2*)(smem_u + XF_LO + dw) = lv;
        }
    }
    __syncthreads();

    const int fl2 = lane * 2;   // uint4 index of this lane's fragment within a block

    // ---------- Phase B: gating logits via WMMA (K split across waves) ----------
    {
        const uint4* wgHi = (const uint4*)(ws + WG_HI);
        const uint4* wgLo = (const uint4*)(ws + WG_LO);
        const uint4* xfHi = (const uint4*)(smem_u + XF_HI);
        const uint4* xfLo = (const uint4*)(smem_u + XF_LO);
        v8f gacc = {};
        for (int kb = wave * (KB_X / 4); kb < (wave + 1) * (KB_X / 4); ++kb) {
            FragU ah, al, bh, bl;
            int fb = kb * 64 + fl2;
            ah.q[0] = xfHi[fb]; ah.q[1] = xfHi[fb + 1];
            al.q[0] = xfLo[fb]; al.q[1] = xfLo[fb + 1];
            bh.q[0] = wgHi[fb]; bh.q[1] = wgHi[fb + 1];
            bl.q[0] = wgLo[fb]; bl.q[1] = wgLo[fb + 1];
            gacc = wmma3(ah, al, bh, bl, gacc);
        }
        float* gpart = (float*)(smem_u + GPART);
        #pragma unroll
        for (int j = 0; j < 8; ++j)
            gpart[wave * 256 + (j + mhigh) * 16 + mn] = gacc[j];
    }
    __syncthreads();

    // ---------- top-2 softmax -> dense cw[16][8] ----------
    if (tid < TILE_M) {
        const float* gpart = (const float*)(smem_u + GPART);
        float* cw = (float*)(smem_u + CW);
        float lg[E_NUM];
        #pragma unroll
        for (int e = 0; e < E_NUM; ++e)
            lg[e] = gpart[tid * 16 + e] + gpart[256 + tid * 16 + e] +
                    gpart[512 + tid * 16 + e] + gpart[768 + tid * 16 + e];
        float best = -3.4e38f; int bi = 0;
        #pragma unroll
        for (int e = 0; e < E_NUM; ++e)
            if (lg[e] > best) { best = lg[e]; bi = e; }
        float second = -3.4e38f; int si = 0;
        #pragma unroll
        for (int e = 0; e < E_NUM; ++e)
            if (e != bi && lg[e] > second) { second = lg[e]; si = e; }
        float t1  = __expf(second - best);
        float inv = 1.0f / (1.0f + t1);
        #pragma unroll
        for (int e = 0; e < E_NUM; ++e)
            cw[tid * 8 + e] = (e == bi) ? inv : ((e == si) ? t1 * inv : 0.0f);
    }
    __syncthreads();

    // ---------- Phase C: h = x @ A (2 experts/wave), scale by cw, split to LDS frags ----------
    {
        const uint4* apHi = (const uint4*)(ws + A_HI);
        const uint4* apLo = (const uint4*)(ws + A_LO);
        const uint4* xfHi = (const uint4*)(smem_u + XF_HI);
        const uint4* xfLo = (const uint4*)(smem_u + XF_LO);
        const int e0 = wave * 2;
        v8f acc0 = {}, acc1 = {};
        #pragma unroll 2
        for (int kb = 0; kb < KB_X; ++kb) {
            FragU ah, al, b0h, b0l, b1h, b1l;
            int fb = kb * 64 + fl2;
            ah.q[0] = xfHi[fb]; ah.q[1] = xfHi[fb + 1];
            al.q[0] = xfLo[fb]; al.q[1] = xfLo[fb + 1];
            int w0 = ((e0 + 0) * KB_X + kb) * 64 + fl2;
            int w1 = ((e0 + 1) * KB_X + kb) * 64 + fl2;
            b0h.q[0] = apHi[w0]; b0h.q[1] = apHi[w0 + 1];
            b0l.q[0] = apLo[w0]; b0l.q[1] = apLo[w0 + 1];
            b1h.q[0] = apHi[w1]; b1h.q[1] = apHi[w1 + 1];
            b1l.q[0] = apLo[w1]; b1l.q[1] = apLo[w1 + 1];
            acc0 = wmma3(ah, al, b0h, b0l, acc0);
            acc1 = wmma3(ah, al, b1h, b1l, acc1);
        }
        // scale by combine weights, hi/lo split, scatter into phase-D A-frag layout
        const float* cw = (const float*)(smem_u + CW);
        unsigned short* hwHi = (unsigned short*)(smem_u + HWF_HI);
        unsigned short* hwLo = (unsigned short*)(smem_u + HWF_LO);
        const int h2 = (mn >> 3) & 1;     // K(local) bit3 comes from r
        #pragma unroll
        for (int j = 0; j < 8; ++j) {
            int m = j + mhigh;
            float v0 = acc0[j] * cw[m * 8 + e0];
            float v1 = acc1[j] * cw[m * 8 + e0 + 1];
            int base = (wave * 32 + m + (h2 << 4)) * 16 + (mn & 7);  // bf16 index
            unsigned int u0 = __float_as_uint(v0);
            hwHi[base] = (unsigned short)(u0 >> 16);
            float rr0 = v0 - __uint_as_float(u0 & 0xffff0000u);
            hwLo[base] = (unsigned short)(__float_as_uint(rr0) >> 16);
            unsigned int u1 = __float_as_uint(v1);
            hwHi[base + 8] = (unsigned short)(u1 >> 16);
            float rr1 = v1 - __uint_as_float(u1 & 0xffff0000u);
            hwLo[base + 8] = (unsigned short)(__float_as_uint(rr1) >> 16);
        }
    }
    __syncthreads();

    // ---------- Phase D: out = hw[16x128] @ B[128x1024], 16 N-tiles per wave ----------
    {
        const uint4* bpHi = (const uint4*)(ws + B_HI);
        const uint4* bpLo = (const uint4*)(ws + B_LO);
        const uint4* hfHi = (const uint4*)(smem_u + HWF_HI);
        const uint4* hfLo = (const uint4*)(smem_u + HWF_LO);
        FragU ah[4], al[4];                     // hoist all K fragments (4 blocks)
        #pragma unroll
        for (int kb = 0; kb < 4; ++kb) {
            int fb = kb * 64 + fl2;
            ah[kb].q[0] = hfHi[fb]; ah[kb].q[1] = hfHi[fb + 1];
            al[kb].q[0] = hfLo[fb]; al[kb].q[1] = hfLo[fb + 1];
        }
        float* og = out + tok0 * D_DIM;
        for (int np = 0; np < 8; ++np) {        // pairs of N-tiles -> 2 indep chains
            int nt0 = wave * 16 + np * 2;
            v8f acc0 = {}, acc1 = {};
            #pragma unroll
            for (int kb = 0; kb < 4; ++kb) {
                FragU b0h, b0l, b1h, b1l;
                int w0 = ((nt0 + 0) * 4 + kb) * 64 + fl2;
                int w1 = ((nt0 + 1) * 4 + kb) * 64 + fl2;
                b0h.q[0] = bpHi[w0]; b0h.q[1] = bpHi[w0 + 1];
                b0l.q[0] = bpLo[w0]; b0l.q[1] = bpLo[w0 + 1];
                b1h.q[0] = bpHi[w1]; b1h.q[1] = bpHi[w1 + 1];
                b1l.q[0] = bpLo[w1]; b1l.q[1] = bpLo[w1 + 1];
                acc0 = wmma3(ah[kb], al[kb], b0h, b0l, acc0);
                acc1 = wmma3(ah[kb], al[kb], b1h, b1l, acc1);
            }
            #pragma unroll
            for (int j = 0; j < 8; ++j) {
                long row = (long)(j + mhigh) * D_DIM;
                og[row + (nt0 + 0) * 16 + mn] = acc0[j];
                og[row + (nt0 + 1) * 16 + mn] = acc1[j];
            }
        }
    }
}

extern "C" void kernel_launch(void* const* d_in, const int* in_sizes, int n_in,
                              void* d_out, int out_size, void* d_ws, size_t ws_size,
                              hipStream_t stream) {
    const float* x  = (const float*)d_in[0];
    const float* Wg = (const float*)d_in[1];
    const float* A  = (const float*)d_in[2];
    const float* Bm = (const float*)d_in[3];
    float* out = (float*)d_out;
    unsigned int* ws = (unsigned int*)d_ws;

    // Pack weights into bf16 hi/lo WMMA fragment planes (L2-resident, ~1 MB)
    pack_weights_kernel<<<dim3(8192 / 256), dim3(256), 0, stream>>>(
        Wg, ws + WG_HI, ws + WG_LO, /*ld=*/E_NUM, /*kblocks=*/32, /*ntiles=*/1, /*nv=*/8);
    pack_weights_kernel<<<dim3(65536 / 256), dim3(256), 0, stream>>>(
        A, ws + A_HI, ws + A_LO, /*ld=*/R_DIM, /*kblocks=*/256, /*ntiles=*/1, /*nv=*/16);
    pack_weights_kernel<<<dim3(65536 / 256), dim3(256), 0, stream>>>(
        Bm, ws + B_HI, ws + B_LO, /*ld=*/D_DIM, /*kblocks=*/4, /*ntiles=*/64, /*nv=*/16);

    const int T = in_sizes[0] / D_DIM;
    molora_bf16x3_kernel<<<dim3(T / TILE_M), dim3(THREADS),
                           (size_t)LDS_DW * sizeof(unsigned int), stream>>>(x, ws, out);
}
